// DAC_558345749225
// MI455X (gfx1250) — matrix operand order repaired and checked by reading, not compiled, not run on hardware
//
#include <hip/hip_runtime.h>
#include <math.h>

// ---------------- types for WMMA ----------------
typedef __attribute__((ext_vector_type(16))) __bf16 v16bf;
typedef __attribute__((ext_vector_type(8)))  float  v8f;

union Frag {
    v16bf v;
    uint4 u[2];
};

// ---------------- problem constants ----------------
#define BB   16
#define CC   256
#define HH   64
#define WW   64
#define GG   8
#define CG   32
#define HID  64
#define ROWS 8          // rows per workgroup in conv kernel

// ---------------- bf16 helpers (manual, storage = ushort) ----------------
__device__ __forceinline__ unsigned short f2bf(float f) {
    union { float f; unsigned int u; } v; v.f = f;
    unsigned int u = v.u;
    unsigned int r = u + 0x7FFFu + ((u >> 16) & 1u);   // round-to-nearest-even
    return (unsigned short)(r >> 16);
}
__device__ __forceinline__ float bf2f(unsigned short h) {
    union { unsigned int u; float f; } v; v.u = ((unsigned int)h) << 16;
    return v.f;
}

// async global->LDS copy of 16 bytes (CDNA5, tracked by ASYNCcnt)
__device__ __forceinline__ void async_copy_b128(unsigned lds_byte_off,
                                                const void* gsrc) {
    asm volatile("global_load_async_to_lds_b128 %0, %1, off"
                 :
                 : "v"(lds_byte_off), "v"((unsigned long long)gsrc)
                 : "memory");
}
__device__ __forceinline__ void wait_asynccnt0() {
    asm volatile("s_wait_asynccnt 0x0" ::: "memory");
}

// =====================================================================
// Kernel 1: x[B,C,H,W] f32 -> xt[b,y,x,c] bf16 (channel-last) + row sums
// One block per (b,y). Coalesced reads, LDS transpose, coalesced writes.
// =====================================================================
#define PADW 264   // 256 + 8 halfs of pad to break LDS bank conflicts

__global__ __launch_bounds__(256)
void k_transpose_pool(const float* __restrict__ x,
                      unsigned short* __restrict__ xt,
                      float* __restrict__ rowsum) {
    __shared__ unsigned short tile[HH * PADW];   // ~33 KB
    const int blk  = blockIdx.x;          // b*64 + y
    const int y    = blk & 63;
    const int b    = blk >> 6;
    const int t    = threadIdx.x;
    const int xcol = t & 63;
    const int coff = t >> 6;              // 0..3

    const float* xin = x + (size_t)b * CC * HH * WW + (size_t)y * WW;
    for (int c0 = 0; c0 < CC; c0 += 4) {
        int c = c0 + coff;
        float v = xin[(size_t)c * HH * WW + xcol];
        tile[xcol * PADW + c] = f2bf(v);
    }
    __syncthreads();

    // per-channel row sum (c = t)
    {
        float s = 0.f;
        for (int xx = 0; xx < WW; ++xx) s += bf2f(tile[xx * PADW + t]);
        rowsum[((size_t)b * HH + y) * CC + t] = s;
    }

    // contiguous bf16 write of this (b,y) slab: [x][c]
    unsigned short* dst = xt + ((size_t)b * HH + y) * (WW * CC);
    for (int i = t; i < (WW * CC) / 2; i += 256) {
        int xi = i >> 7;              // / 128 pairs
        int c2 = (i & 127) << 1;
        unsigned int lo = tile[xi * PADW + c2];
        unsigned int hi = tile[xi * PADW + c2 + 1];
        *(unsigned int*)(dst + xi * CC + c2) = lo | (hi << 16);
    }
}

// =====================================================================
// Kernel 2: pooled[b,c] = mean over H,W via row sums
// =====================================================================
__global__ __launch_bounds__(256)
void k_pool_reduce(const float* __restrict__ rowsum, float* __restrict__ pooled) {
    int idx = blockIdx.x * 256 + threadIdx.x;    // b*C + c  (4096 total)
    int b = idx >> 8, c = idx & 255;
    float s = 0.f;
    for (int yy = 0; yy < HH; ++yy) s += rowsum[((size_t)b * HH + yy) * CC + c];
    pooled[idx] = s * (1.0f / (HH * WW));
}

// =====================================================================
// Kernel 3: both tiny MLPs in one block.
// =====================================================================
__global__ __launch_bounds__(256)
void k_mlp(const float* __restrict__ pooled,
           const float* __restrict__ ap_w1, const float* __restrict__ ap_b1,
           const float* __restrict__ ap_w2, const float* __restrict__ ap_b2,
           const float* __restrict__ mw1,  const float* __restrict__ mb1,
           const float* __restrict__ mw2,  const float* __restrict__ mb2,
           float* __restrict__ angles, float* __restrict__ mod) {
    __shared__ float pool_s[BB * CC];
    __shared__ float h_s[BB * HID];
    __shared__ float m_s[BB * HID];
    const int t = threadIdx.x;
    for (int i = t; i < BB * CC; i += 256) pool_s[i] = pooled[i];
    __syncthreads();

    for (int i = t; i < BB * HID; i += 256) {
        int b = i >> 6, j = i & 63;
        float sa = ap_b1[j], sm = mb1[j];
        for (int c = 0; c < CC; ++c) {
            float pv = pool_s[b * CC + c];
            sa += pv * ap_w1[j * CC + c];
            sm += pv * mw1[j * CC + c];
        }
        h_s[i] = fmaxf(sa, 0.f);
        m_s[i] = fmaxf(sm, 0.f);
    }
    __syncthreads();

    if (t < BB * GG) {
        int b = t >> 3, gq = t & 7;
        float s = ap_b2[gq];
        for (int j = 0; j < HID; ++j) s += h_s[b * HID + j] * ap_w2[gq * HID + j];
        angles[t] = tanhf(s) * 0.78539816339744830961f;  // pi/4
    }
    for (int i = t; i < BB * CC; i += 256) {
        int b = i >> 8, c = i & 255;
        float s = mb2[c];
        for (int j = 0; j < HID; ++j) s += m_s[b * HID + j] * mw2[c * HID + j];
        mod[i] = 1.f / (1.f + expf(-s));
    }
}

// =====================================================================
// Kernel 4: bilinear-rotate base kernels -> bf16 wrot[bg][pos][co][ci]
// =====================================================================
__global__ __launch_bounds__(256)
void k_rotate(const float* __restrict__ base, const float* __restrict__ angles,
              unsigned short* __restrict__ wrot) {
    const int bg = blockIdx.x;          // b*8 + g
    const int g  = bg & 7;
    const float ang = angles[bg];
    const float cth = cosf(ang), sth = sinf(ang);
    const float* kb = base + (size_t)g * CG * CG * 9;
    unsigned short* wo = wrot + (size_t)bg * (9 * CG * CG);

    for (int i = threadIdx.x; i < 9 * CG * CG; i += 256) {
        int p   = i >> 10;          // 0..8
        int rem = i & 1023;
        int co  = rem >> 5;
        int ci  = rem & 31;
        int ky  = p / 3, kx = p - 3 * (p / 3);
        float xx = (float)kx - 1.0f, yy = (float)ky - 1.0f;
        float xr =  xx * cth + yy * sth + 1.0f;
        float yr = -xx * sth + yy * cth + 1.0f;
        float x0 = fminf(fmaxf(floorf(xr), 0.f), 1.f);
        float y0 = fminf(fmaxf(floorf(yr), 0.f), 1.f);
        float x1 = x0 + 1.f, y1 = y0 + 1.f;
        float w00 = (x1 - xr) * (y1 - yr);
        float w01 = (x1 - xr) * (yr - y0);
        float w10 = (xr - x0) * (y1 - yr);
        float w11 = (xr - x0) * (yr - y0);
        int x0i = (int)x0, y0i = (int)y0;
        const float* kk = kb + ((size_t)co * CG + ci) * 9;
        float v = w00 * kk[y0i * 3 + x0i]       + w01 * kk[(y0i + 1) * 3 + x0i]
                + w10 * kk[y0i * 3 + x0i + 1]   + w11 * kk[(y0i + 1) * 3 + x0i + 1];
        wo[i] = f2bf(v);
    }
}

// =====================================================================
// Kernel 5: shift-GEMM grouped conv on WMMA + fused SE gate.
// Grid: (b,g,rowblock) = 16*8*8 = 1024 blocks, 8 waves/block.
// Activations (10 rows with y-halo) and rotated weights staged to LDS
// with CDNA5 async copies (ASYNCcnt); fragments fetched with ds_load_b128.
// Wave: one image row (64 px) x 32 co = 2 M-tiles x 4 N-tiles,
//       9 taps x K=32 -> 72 v_wmma_f32_16x16x32_bf16 per wave.
// =====================================================================
__global__ __launch_bounds__(256)
void k_conv_wmma(const unsigned short* __restrict__ xt,
                 const unsigned short* __restrict__ wrot,
                 const float* __restrict__ mod,
                 float* __restrict__ out) {
    __shared__ __align__(16) unsigned short lds_w[9 * CG * CG];          // 18 KB
    __shared__ __align__(16) unsigned short lds_x[(ROWS + 2) * WW * CG]; // 40 KB
    const int blk    = blockIdx.x;
    const int rowblk = blk & 7;            // H/ROWS = 8
    const int bg     = blk >> 3;
    const int g      = bg & 7;
    const int b      = bg >> 3;
    const int t      = threadIdx.x;
    const int y0     = rowblk * ROWS;

    const unsigned short* xb = xt + (size_t)b * HH * WW * CC;

    // ---- async stage: rotated weights (1152 x b128) ----
    {
        const unsigned short* src = wrot + (size_t)bg * (9 * CG * CG);
        unsigned wbase = (unsigned)(unsigned long long)(&lds_w[0]);
        for (int i = t; i < (9 * CG * CG) / 8; i += 256)
            async_copy_b128(wbase + i * 16, src + i * 8);
    }
    // ---- async stage: 10 activation rows (y0-1 .. y0+8), clamped ----
    {
        unsigned xbase_lds = (unsigned)(unsigned long long)(&lds_x[0]);
        const int px = t >> 2;             // 0..63
        const int cq = (t & 3) * 8;        // 0,8,16,24
        for (int r = 0; r < ROWS + 2; ++r) {
            int ysrc = y0 - 1 + r;
            int yc   = min(max(ysrc, 0), HH - 1);   // clamped; halo rows zeroed later
            const unsigned short* src =
                xb + ((size_t)yc * WW + px) * CC + g * CG + cq;
            unsigned dst = xbase_lds + (((r * WW + px) * CG) + cq) * 2;
            async_copy_b128(dst, src);
        }
    }
    wait_asynccnt0();
    __syncthreads();

    const int wv    = t >> 5;
    const int lane  = t & 31;
    const int l15   = lane & 15;
    const int lhalf = lane >> 4;
    const int y     = y0 + wv;

    // per-register SE gate factors (C/D layout: M = r + 8*lane_half)
    float modf[2][8];
#pragma unroll
    for (int mt = 0; mt < 2; ++mt)
#pragma unroll
        for (int r = 0; r < 8; ++r)
            modf[mt][r] = mod[b * CC + g * CG + mt * 16 + r + 8 * lhalf];

    v8f acc[4][2];
#pragma unroll
    for (int nt = 0; nt < 4; ++nt)
#pragma unroll
        for (int mt = 0; mt < 2; ++mt) { v8f z = {}; acc[nt][mt] = z; }

#pragma unroll
    for (int p = 0; p < 9; ++p) {
        const int dy = p / 3 - 1, dx = p % 3 - 1;
        const int ysrc = y + dy;
        const int lrow = wv + dy + 1;                 // 0..9 in lds_x
        const bool yok = (ysrc >= 0) && (ysrc < HH);

        // A fragments from LDS (lane m, K = {h*8.., 16+h*8..})
        Frag a[2];
#pragma unroll
        for (int mt = 0; mt < 2; ++mt) {
            const unsigned short* wp =
                lds_w + ((p * CG) + (mt * 16 + l15)) * CG + lhalf * 8;
            a[mt].u[0] = *(const uint4*)(wp);
            a[mt].u[1] = *(const uint4*)(wp + 16);
        }

#pragma unroll
        for (int nt = 0; nt < 4; ++nt) {
            const int xp = nt * 16 + l15 + dx;
            const int xc = min(max(xp, 0), WW - 1);
            const bool ok = yok && (xp >= 0) && (xp < WW);
            const unsigned short* bp =
                lds_x + ((lrow * WW + xc) * CG) + lhalf * 8;
            uint4 b0 = *(const uint4*)(bp);
            uint4 b1 = *(const uint4*)(bp + 16);
            uint4 zz = {0u, 0u, 0u, 0u};
            Frag bf;
            bf.u[0] = ok ? b0 : zz;      // lane-level select; EXEC stays all-ones
            bf.u[1] = ok ? b1 : zz;
            acc[nt][0] = __builtin_amdgcn_wmma_f32_16x16x32_bf16(
                false, a[0].v, false, bf.v, (short)0, acc[nt][0], false, false);
            acc[nt][1] = __builtin_amdgcn_wmma_f32_16x16x32_bf16(
                false, a[1].v, false, bf.v, (short)0, acc[nt][1], false, false);
        }
    }

    // store with fused gate: out[b, co, y, x] = acc * mod[b, co]
#pragma unroll
    for (int nt = 0; nt < 4; ++nt) {
        const int xo = nt * 16 + l15;
#pragma unroll
        for (int mt = 0; mt < 2; ++mt) {
#pragma unroll
            for (int r = 0; r < 8; ++r) {
                const int co = g * CG + mt * 16 + r + 8 * lhalf;
                out[(((size_t)b * CC + co) * HH + y) * WW + xo] =
                    acc[nt][mt][r] * modf[mt][r];
            }
        }
    }
}

// =====================================================================
// Host launcher
// =====================================================================
extern "C" void kernel_launch(void* const* d_in, const int* in_sizes, int n_in,
                              void* d_out, int out_size, void* d_ws, size_t ws_size,
                              hipStream_t stream) {
    const float* x           = (const float*)d_in[0];
    const float* base_kernel = (const float*)d_in[1];
    const float* ap_w1       = (const float*)d_in[2];
    const float* ap_b1       = (const float*)d_in[3];
    const float* ap_w2       = (const float*)d_in[4];
    const float* ap_b2       = (const float*)d_in[5];
    const float* meta_w1     = (const float*)d_in[6];
    const float* meta_b1     = (const float*)d_in[7];
    const float* meta_w2     = (const float*)d_in[8];
    const float* meta_b2     = (const float*)d_in[9];
    float* out = (float*)d_out;

    // workspace layout (bytes, 256-aligned blocks)
    char* ws = (char*)d_ws;
    const size_t xt_bytes     = (size_t)BB * HH * WW * CC * 2;      // 32 MB
    const size_t wrot_bytes   = (size_t)BB * GG * 9 * CG * CG * 2;  // 2.25 MB
    const size_t rowsum_bytes = (size_t)BB * HH * CC * 4;           // 1 MB
    const size_t pooled_bytes = (size_t)BB * CC * 4;

    unsigned short* xt   = (unsigned short*)(ws);
    unsigned short* wrot = (unsigned short*)(ws + xt_bytes);
    float* rowsum        = (float*)(ws + xt_bytes + wrot_bytes);
    float* pooled        = (float*)(ws + xt_bytes + wrot_bytes + rowsum_bytes);
    float* angles        = (float*)(ws + xt_bytes + wrot_bytes + rowsum_bytes + pooled_bytes);
    float* mod           = (float*)(ws + xt_bytes + wrot_bytes + rowsum_bytes + pooled_bytes + 512);

    k_transpose_pool<<<BB * HH, 256, 0, stream>>>(x, xt, rowsum);
    k_pool_reduce<<<(BB * CC) / 256, 256, 0, stream>>>(rowsum, pooled);
    k_mlp<<<1, 256, 0, stream>>>(pooled, ap_w1, ap_b1, ap_w2, ap_b2,
                                 meta_w1, meta_b1, meta_w2, meta_b2, angles, mod);
    k_rotate<<<BB * GG, 256, 0, stream>>>(base_kernel, angles, wrot);
    k_conv_wmma<<<BB * GG * (HH / ROWS), 256, 0, stream>>>(xt, wrot, mod, out);
}
